// CrossAttention_89446988906571
// MI455X (gfx1250) — matrix operand order repaired and checked by reading, not compiled, not run on hardware
//
#include <hip/hip_runtime.h>
#include <hip/hip_bf16.h>

// ---------------------------------------------------------------------------
// Types for CDNA5 WMMA (gfx1250, wave32)
// ---------------------------------------------------------------------------
typedef __attribute__((ext_vector_type(16))) __bf16 v16bf;
typedef __attribute__((ext_vector_type(8)))  float  v8f;
typedef __attribute__((ext_vector_type(4)))  unsigned v4u;
typedef __attribute__((ext_vector_type(8)))  int      v8i_t;
typedef __attribute__((ext_vector_type(4)))  int      v4i_t;

union Frag {
    v16bf    v;
    unsigned u[8];
};

__device__ __forceinline__ unsigned short f2bf(float f) {
    unsigned u = __float_as_uint(f);
    unsigned r = u + 0x7FFFu + ((u >> 16) & 1u);   // round-to-nearest-even
    return (unsigned short)(r >> 16);
}

// Constants of this problem
#define DD   1024
#define LL   768
#define HH   16
#define HD   64
#define BB   4
#define TT   4096
#define NN   512

// TDM availability (this toolchain: 6-arg form, last vector arg is int32x8)
#if defined(__has_builtin)
#  if __has_builtin(__builtin_amdgcn_tensor_load_to_lds) && \
      __has_builtin(__builtin_amdgcn_s_wait_tensorcnt)
#    define USE_TDM 1
#  endif
#endif

// ---------------------------------------------------------------------------
// Kernel 1: LayerNorm (fp32 in) -> bf16 out.  One block per row.
// ---------------------------------------------------------------------------
__global__ __launch_bounds__(256)
void ln_to_bf16_kernel(const float* __restrict__ in,
                       const float* __restrict__ gamma,
                       const float* __restrict__ beta,
                       unsigned short* __restrict__ out,
                       int cols) {
    const int row = blockIdx.x;
    const float* r = in + (size_t)row * cols;

    float s = 0.f, s2 = 0.f;
    for (int c = threadIdx.x; c < cols; c += 256) {
        float v = r[c];
        s  += v;
        s2 += v * v;
    }

    __shared__ float red[256];
    red[threadIdx.x] = s;  __syncthreads();
    for (int off = 128; off > 0; off >>= 1) {
        if (threadIdx.x < off) red[threadIdx.x] += red[threadIdx.x + off];
        __syncthreads();
    }
    const float mean = red[0] / (float)cols;
    __syncthreads();
    red[threadIdx.x] = s2; __syncthreads();
    for (int off = 128; off > 0; off >>= 1) {
        if (threadIdx.x < off) red[threadIdx.x] += red[threadIdx.x + off];
        __syncthreads();
    }
    const float var = red[0] / (float)cols - mean * mean;
    const float inv = rsqrtf(var + 1e-5f);

    unsigned short* orow = out + (size_t)row * cols;
    for (int c = threadIdx.x; c < cols; c += 256) {
        float v = (r[c] - mean) * inv * gamma[c] + beta[c];
        orow[c] = f2bf(v);
    }
}

// ---------------------------------------------------------------------------
// Kernel 2: W[k][n] fp32 (KxN, row-major) -> Wt[n][k] bf16 (NxK, row-major)
// ---------------------------------------------------------------------------
__global__ __launch_bounds__(256)
void transpose_to_bf16_kernel(const float* __restrict__ W,
                              unsigned short* __restrict__ Wt,
                              int K, int N) {
    size_t i = (size_t)blockIdx.x * 256 + threadIdx.x;   // over K*N
    int k = (int)(i / N);
    int n = (int)(i % N);
    Wt[(size_t)n * K + k] = f2bf(W[i]);
}

// ---------------------------------------------------------------------------
// Kernel 3: LDS-tiled WMMA GEMM.
//  C[M,N](bf16) = A[M,K](bf16) @ Wt[N,K](bf16)^T + bias, f32 accumulate.
//  Block (8 waves) computes a 128x128 tile; per K-step stages A(128x32) and
//  B(128x32) into LDS (16 KB); wave w computes the 16x128 strip at rows w*16,
//  reusing one A-fragment across 8 WMMAs (~256 B global traffic per WMMA).
// ---------------------------------------------------------------------------
__global__ __launch_bounds__(256)
void gemm_bf16_wmma_kernel(const unsigned short* __restrict__ A,
                           const unsigned short* __restrict__ Wt,
                           const float* __restrict__ bias,
                           unsigned short* __restrict__ C,
                           int M, int N, int K) {
    __shared__ unsigned short As[128 * 32];
    __shared__ unsigned short Bs[128 * 32];

    const int w    = threadIdx.x >> 5;
    const int lane = threadIdx.x & 31;
    const int g    = lane >> 4;
    const int c    = lane & 15;

    const int nb = N >> 7;                   // 128-wide column blocks
    const int m0 = (blockIdx.x / nb) * 128;
    const int n0 = (blockIdx.x % nb) * 128;

    v8f acc[8];
    #pragma unroll
    for (int t = 0; t < 8; ++t)
        #pragma unroll
        for (int r = 0; r < 8; ++r) acc[t][r] = 0.f;

    for (int k0 = 0; k0 < K; k0 += 32) {
        // ---- stage A/B K-slices (16B vector copies, fully coalesced) ----
        #pragma unroll
        for (int idx = threadIdx.x; idx < 512; idx += 256) {
            int row = idx >> 2, part = (idx & 3) * 8;
            *(uint4*)(&As[row * 32 + part]) =
                *(const uint4*)(A  + (size_t)(m0 + row) * K + k0 + part);
            *(uint4*)(&Bs[row * 32 + part]) =
                *(const uint4*)(Wt + (size_t)(n0 + row) * K + k0 + part);
        }
        __syncthreads();

        // ---- A fragment for this wave's 16-row strip ----
        Frag a;
        const unsigned short* ar = &As[(w * 16 + c) * 32];
        #pragma unroll
        for (int i = 0; i < 4; ++i) {
            a.u[i]     = *(const unsigned*)(ar + g * 16 +     2 * i);
            a.u[4 + i] = *(const unsigned*)(ar + g * 16 + 8 + 2 * i);
        }

        // ---- 8 column tiles: reuse A-frag across 8 WMMAs ----
        #pragma unroll
        for (int t = 0; t < 8; ++t) {
            Frag bfr;
            const unsigned short* br = &Bs[(t * 16 + c) * 32];
            #pragma unroll
            for (int i = 0; i < 8; ++i)
                bfr.u[i] = *(const unsigned*)(br + g * 16 + 2 * i);
            acc[t] = __builtin_amdgcn_wmma_f32_16x16x32_bf16(
                         false, a.v, false, bfr.v, (short)0, acc[t], false, false);
        }
        __syncthreads();
    }

    // ---- epilogue: bias + bf16 store ----
    #pragma unroll
    for (int t = 0; t < 8; ++t) {
        const float bv = bias[n0 + t * 16 + c];
        #pragma unroll
        for (int r = 0; r < 8; ++r) {
            int row = m0 + w * 16 + r + 8 * g;
            C[(size_t)row * N + n0 + t * 16 + c] = f2bf(acc[t][r] + bv);
        }
    }
}

// ---------------------------------------------------------------------------
// Kernel 4: flash cross-attention.
//  grid = (T/128, B*H), block = 256 (8 waves); each wave owns a 16-row Q tile.
//  K chunk staged into LDS by the Tensor Data Mover (TDM); V chunk staged
//  transposed by VALU stores; online softmax in fp32.
// ---------------------------------------------------------------------------
__global__ __launch_bounds__(256)
void flash_attn_wmma_kernel(const unsigned short* __restrict__ q,
                            const unsigned short* __restrict__ k,
                            const unsigned short* __restrict__ v,
                            float* __restrict__ out) {
    __shared__ unsigned short Ks[64 * 64];     // [n_local][hd]
    __shared__ unsigned short Vt[64 * 64];     // [hd][n_local]  (transposed)
    __shared__ unsigned short Pl[8][16 * 32];  // per-wave P relayout scratch

    const int b    = blockIdx.y >> 4;          // H = 16
    const int h    = blockIdx.y & 15;
    const int w    = threadIdx.x >> 5;
    const int lane = threadIdx.x & 31;
    const int g    = lane >> 4;
    const int c    = lane & 15;
    const int t0   = blockIdx.x * 128 + w * 16;

    // Q A-fragments (held in registers for the whole kernel): hd 0..31 / 32..63
    const unsigned short* qrow =
        q + ((size_t)(b * TT + t0 + c)) * DD + h * HD;
    Frag a_lo, a_hi;
    #pragma unroll
    for (int i = 0; i < 4; ++i) {
        a_lo.u[i]     = *(const unsigned*)(qrow + g * 16 +     2 * i);
        a_lo.u[4 + i] = *(const unsigned*)(qrow + g * 16 + 8 + 2 * i);
        a_hi.u[i]     = *(const unsigned*)(qrow + 32 + g * 16 +     2 * i);
        a_hi.u[4 + i] = *(const unsigned*)(qrow + 32 + g * 16 + 8 + 2 * i);
    }

    float m_r[8], l_r[8];
    v8f o[4];
    #pragma unroll
    for (int r = 0; r < 8; ++r) { m_r[r] = -1e30f; l_r[r] = 0.f; }
    #pragma unroll
    for (int f = 0; f < 4; ++f)
        #pragma unroll
        for (int r = 0; r < 8; ++r) o[f][r] = 0.f;

    for (int nc = 0; nc < NN; nc += 64) {
#ifdef USE_TDM
        // ---- K chunk: 2D tile copy (64 rows x 128 B, stride D) via TDM ----
        if (w == 0) {
            unsigned long long gaddr = (unsigned long long)(const void*)
                (k + ((size_t)(b * NN + nc)) * DD + h * HD);
            unsigned ldsoff = (unsigned)(unsigned long long)(const void*)&Ks[0];
            // D# group0: count=1 | lds_addr | global_addr(57b) | type=2
            v4u g0 = { 1u,
                       ldsoff,
                       (unsigned)gaddr,
                       (unsigned)((gaddr >> 32) & 0x01FFFFFFu) | 0x80000000u };
            // D# group1: data_size=2B, tensor_dim0=64, tensor_dim1=64,
            //            tile_dim0=64, tile_dim1=64, dim0_stride=1024
            v8i_t g1 = { (int)0x00010000, (int)(64 << 16), (int)(64 << 16),
                         (int)(64 << 16), 64, 1024, 0, 0 };
            v4i_t gz4 = { 0, 0, 0, 0 };
            v8i_t gz8 = { 0, 0, 0, 0, 0, 0, 0, 0 };
            __builtin_amdgcn_tensor_load_to_lds(g0, g1, gz4, gz4, gz8, 0);
        }
        // ---- V chunk: transposed staging by all 8 waves ----
        for (int idx = threadIdx.x; idx < 64 * 64; idx += 256) {
            int nl = idx >> 6, d = idx & 63;
            Vt[d * 64 + nl] = v[((size_t)(b * NN + nc + nl)) * DD + h * HD + d];
        }
        if (w == 0) __builtin_amdgcn_s_wait_tensorcnt(0);
        __syncthreads();
#else
        for (int idx = threadIdx.x; idx < 64 * 64; idx += 256) {
            int nl = idx >> 6, d = idx & 63;
            size_t gidx = ((size_t)(b * NN + nc + nl)) * DD + h * HD + d;
            Ks[idx]         = k[gidx];
            Vt[d * 64 + nl] = v[gidx];
        }
        __syncthreads();
#endif

        #pragma unroll
        for (int s = 0; s < 2; ++s) {            // two 32-column sub-tiles
            const int ns = s * 32;

            // ---- S = (Q K^T) * 1/sqrt(hd) : two 16-col score tiles ----
            Frag b0l, b0h, b1l, b1h;
            #pragma unroll
            for (int i = 0; i < 8; ++i) {
                b0l.u[i] = *(const unsigned*)(&Ks[(ns + c)      * 64 +      g * 16 + 2 * i]);
                b0h.u[i] = *(const unsigned*)(&Ks[(ns + c)      * 64 + 32 + g * 16 + 2 * i]);
                b1l.u[i] = *(const unsigned*)(&Ks[(ns + c + 16) * 64 +      g * 16 + 2 * i]);
                b1h.u[i] = *(const unsigned*)(&Ks[(ns + c + 16) * 64 + 32 + g * 16 + 2 * i]);
            }
            v8f s0 = {0.f,0.f,0.f,0.f,0.f,0.f,0.f,0.f};
            v8f s1 = {0.f,0.f,0.f,0.f,0.f,0.f,0.f,0.f};
            s0 = __builtin_amdgcn_wmma_f32_16x16x32_bf16(false, a_lo.v, false, b0l.v, (short)0, s0, false, false);
            s0 = __builtin_amdgcn_wmma_f32_16x16x32_bf16(false, a_hi.v, false, b0h.v, (short)0, s0, false, false);
            s1 = __builtin_amdgcn_wmma_f32_16x16x32_bf16(false, a_lo.v, false, b1l.v, (short)0, s1, false, false);
            s1 = __builtin_amdgcn_wmma_f32_16x16x32_bf16(false, a_hi.v, false, b1h.v, (short)0, s1, false, false);

            // ---- online softmax update (row reductions across 16-lane halves)
            float al[8];
            #pragma unroll
            for (int r = 0; r < 8; ++r) {
                float v0 = s0[r] * 0.125f;       // 1/sqrt(64)
                float v1 = s1[r] * 0.125f;
                float mx = fmaxf(v0, v1);
                mx = fmaxf(mx, __shfl_xor(mx, 1, 32));
                mx = fmaxf(mx, __shfl_xor(mx, 2, 32));
                mx = fmaxf(mx, __shfl_xor(mx, 4, 32));
                mx = fmaxf(mx, __shfl_xor(mx, 8, 32));
                float mn = fmaxf(m_r[r], mx);
                al[r]    = __expf(m_r[r] - mn);
                m_r[r]   = mn;
                float p0 = __expf(v0 - mn);
                float p1 = __expf(v1 - mn);
                s0[r] = p0; s1[r] = p1;
                float rs = p0 + p1;
                rs += __shfl_xor(rs, 1, 32);
                rs += __shfl_xor(rs, 2, 32);
                rs += __shfl_xor(rs, 4, 32);
                rs += __shfl_xor(rs, 8, 32);
                l_r[r] = l_r[r] * al[r] + rs;
            }
            #pragma unroll
            for (int f = 0; f < 4; ++f)
                #pragma unroll
                for (int r = 0; r < 8; ++r) o[f][r] *= al[r];

            // ---- relayout P (C-frag) -> A-frag via per-wave LDS scratch ----
            unsigned short* pw = &Pl[w][0];
            #pragma unroll
            for (int r = 0; r < 8; ++r) {
                int row = 8 * g + r;
                pw[row * 32 + c]      = f2bf(s0[r]);
                pw[row * 32 + c + 16] = f2bf(s1[r]);
            }
            Frag pa;
            #pragma unroll
            for (int i = 0; i < 4; ++i) {
                pa.u[i]     = *(const unsigned*)(pw + c * 32 + g * 16 +     2 * i);
                pa.u[4 + i] = *(const unsigned*)(pw + c * 32 + g * 16 + 8 + 2 * i);
            }

            // ---- O += P @ V : four 16-col output fragments ----
            #pragma unroll
            for (int f = 0; f < 4; ++f) {
                Frag vb;
                #pragma unroll
                for (int i = 0; i < 8; ++i)
                    vb.u[i] = *(const unsigned*)(&Vt[(f * 16 + c) * 64 + ns + g * 16 + 2 * i]);
                o[f] = __builtin_amdgcn_wmma_f32_16x16x32_bf16(
                           false, pa.v, false, vb.v, (short)0, o[f], false, false);
            }
        }
        __syncthreads();
    }

    // ---- epilogue: normalize and store fp32 ----
    #pragma unroll
    for (int r = 0; r < 8; ++r) {
        int   row  = t0 + 8 * g + r;
        float invl = 1.f / l_r[r];
        float* orow = out + ((size_t)(b * TT + row)) * DD + h * HD;
        #pragma unroll
        for (int f = 0; f < 4; ++f)
            orow[f * 16 + c] = o[f][r] * invl;
    }
}

// ---------------------------------------------------------------------------
// Host launcher
// ---------------------------------------------------------------------------
extern "C" void kernel_launch(void* const* d_in, const int* in_sizes, int n_in,
                              void* d_out, int out_size, void* d_ws, size_t ws_size,
                              hipStream_t stream) {
    const float* x       = (const float*)d_in[0];
    const float* xf      = (const float*)d_in[1];
    const float* norm_g  = (const float*)d_in[2];
    const float* norm_b  = (const float*)d_in[3];
    const float* tnorm_g = (const float*)d_in[4];
    const float* tnorm_b = (const float*)d_in[5];
    const float* Wq      = (const float*)d_in[6];
    const float* bq      = (const float*)d_in[7];
    const float* Wk      = (const float*)d_in[8];
    const float* bk      = (const float*)d_in[9];
    const float* Wv      = (const float*)d_in[10];
    const float* bv      = (const float*)d_in[11];
    float* out = (float*)d_out;

    // workspace carve-out (bf16 buffers as ushort)
    char* ws = (char*)d_ws;
    size_t off = 0;
    auto carve = [&](size_t elems) {
        unsigned short* p = (unsigned short*)(ws + off);
        off += elems * sizeof(unsigned short);
        return p;
    };
    unsigned short* xn   = carve((size_t)BB * TT * DD);
    unsigned short* xfn  = carve((size_t)BB * NN * LL);
    unsigned short* Wq_t = carve((size_t)DD * DD);
    unsigned short* Wk_t = carve((size_t)DD * LL);
    unsigned short* Wv_t = carve((size_t)DD * LL);
    unsigned short* qb   = carve((size_t)BB * TT * DD);
    unsigned short* kb   = carve((size_t)BB * NN * DD);
    unsigned short* vb   = carve((size_t)BB * NN * DD);

    // 1. LayerNorms -> bf16
    ln_to_bf16_kernel<<<BB * TT, 256, 0, stream>>>(x, norm_g, norm_b, xn, DD);
    ln_to_bf16_kernel<<<BB * NN, 256, 0, stream>>>(xf, tnorm_g, tnorm_b, xfn, LL);

    // 2. Weight transpose + bf16 convert
    transpose_to_bf16_kernel<<<(DD * DD) / 256, 256, 0, stream>>>(Wq, Wq_t, DD, DD);
    transpose_to_bf16_kernel<<<(LL * DD) / 256, 256, 0, stream>>>(Wk, Wk_t, LL, DD);
    transpose_to_bf16_kernel<<<(LL * DD) / 256, 256, 0, stream>>>(Wv, Wv_t, LL, DD);

    // 3. Projections (LDS-tiled WMMA GEMM): 128x128 tile per block
    gemm_bf16_wmma_kernel<<<(BB * TT / 128) * (DD / 128), 256, 0, stream>>>(
        xn, Wq_t, bq, qb, BB * TT, DD, DD);
    gemm_bf16_wmma_kernel<<<(BB * NN / 128) * (DD / 128), 256, 0, stream>>>(
        xfn, Wk_t, bk, kb, BB * NN, DD, LL);
    gemm_bf16_wmma_kernel<<<(BB * NN / 128) * (DD / 128), 256, 0, stream>>>(
        xfn, Wv_t, bv, vb, BB * NN, DD, LL);

    // 4. Flash attention (WMMA QK^T and PV, TDM K staging, online softmax)
    dim3 agrid(TT / 128, BB * HH);
    flash_attn_wmma_kernel<<<agrid, 256, 0, stream>>>(qb, kb, vb, out);
}
